// CrossAttention_7378753814895
// MI455X (gfx1250) — matrix operand order repaired and checked by reading, not compiled, not run on hardware
//
#include <hip/hip_runtime.h>

// ---------------------------------------------------------------------------
// CDNA5 (gfx1250) cross-attention: v_wmma_f32_16x16x32_bf16 everywhere,
// flash-softmax with transposed score tiles (softmax rows lane-resident),
// TDM tensor_load_to_lds for V tiles (TENSORcnt), async global->LDS staging
// in the combine kernel (ASYNCcnt), global prefetch, pipelined K fragments.
// ---------------------------------------------------------------------------

typedef __attribute__((ext_vector_type(16))) __bf16 v16bf;
typedef __attribute__((ext_vector_type(8)))  __bf16 v8bf;
typedef __attribute__((ext_vector_type(8)))  float  v8f;
typedef __attribute__((ext_vector_type(4)))  unsigned int u32x4;
typedef __attribute__((ext_vector_type(8)))  int i32x8;
typedef __attribute__((ext_vector_type(4)))  int i32x4;

#define BB   4
#define CC   256
#define CQKD 32
#define NN   4096

// round-to-nearest-even f32 -> bf16 without relying on __bf16 arithmetic cvt
static __device__ __forceinline__ __bf16 f2bf(float f) {
    union { float f; unsigned u; } x; x.f = f;
    unsigned r = x.u + 0x7FFFu + ((x.u >> 16) & 1u);
    unsigned short hs = (unsigned short)(r >> 16);
    __bf16 out;
    __builtin_memcpy(&out, &hs, 2);
    return out;
}

static __device__ __forceinline__ v16bf frag_from_contig(const __bf16* p0, const __bf16* p1) {
    v8bf lo = *(const v8bf*)p0;
    v8bf hi = *(const v8bf*)p1;
    v16bf r;
#pragma unroll
    for (int t = 0; t < 8; ++t) { r[t] = lo[t]; r[8 + t] = hi[t]; }
    return r;
}

// A operand (16 x 32 tile): element [m][k], row-major, leading dim ld.
// Lane L: row m = L%16; h = L/16 selects K {16h..16h+15}.
static __device__ __forceinline__ v16bf load_fragA(const __bf16* base, int ld, int row,
                                                   int k0, int lane) {
    int h = (lane >> 4) & 1;
    const __bf16* p = base + (size_t)row * ld + k0;
    return frag_from_contig(p + 16 * h, p + 16 * h + 8);
}

// B operand (32 x 16 tile): element [k][col] at base + col*ld + k (K contiguous).
static __device__ __forceinline__ v16bf load_fragB(const __bf16* base, int ld, int col,
                                                   int k0, int lane) {
    int kh = (lane >> 4) & 1;
    const __bf16* p = base + (size_t)col * ld + k0 + 16 * kh;
    return frag_from_contig(p, p + 8);
}

static __device__ __forceinline__ v8f wmma_bf16(v16bf a, v16bf b, v8f c) {
    return __builtin_amdgcn_wmma_f32_16x16x32_bf16(false, a, false, b, (short)0, c,
                                                   false, false);
}

// ---------------------------------------------------------------------------
// TDM: load a [CC rows x 32 cols] bf16 tile (rows strided by NN elements in
// global memory, packed contiguously in LDS) via tensor_load_to_lds.
// D# per CDNA5 ISA ch.8: group0 = control/lds/global addr, group1 = dims.
// 6-arg builtin form (clang-23 / therock headers).
// ---------------------------------------------------------------------------
static __device__ __forceinline__ void tdm_load_v_tile(unsigned lds_off,
                                                       const __bf16* gaddr) {
    unsigned long long ga = (unsigned long long)(size_t)gaddr;
    u32x4 g0;
    g0[0] = 1u;                                              // count=1 (valid user D#)
    g0[1] = lds_off;                                         // lds_addr (bytes)
    g0[2] = (unsigned)(ga & 0xFFFFFFFFu);                    // global_addr[31:0]
    g0[3] = (unsigned)((ga >> 32) & 0x1FFFFFFu) | (2u << 30);// addr[56:32] | type=2
    i32x8 g1;
    g1[0] = (int)(1u << 16);                                 // data_size = 2 bytes
    g1[1] = (int)(((unsigned)NN & 0xFFFFu) << 16);           // tensor_dim0 lo16
    g1[2] = (int)((((unsigned)NN >> 16) & 0xFFFFu) |
                  (((unsigned)CC & 0xFFFFu) << 16));         // dim0 hi16 | dim1 lo16
    g1[3] = (int)((((unsigned)CC >> 16) & 0xFFFFu) |
                  (32u << 16));                              // dim1 hi16 | tile_dim0=32
    g1[4] = (int)((unsigned)CC & 0xFFFFu);                   // tile_dim1=256, tile_dim2=0
    g1[5] = (int)(unsigned)NN;                               // tensor_dim0_stride lo32
    g1[6] = 0;                                               // stride0 hi | stride1 lo
    g1[7] = 0;
    i32x4 g2 = {0, 0, 0, 0};
    i32x4 g3 = {0, 0, 0, 0};
    i32x8 g4 = {0, 0, 0, 0, 0, 0, 0, 0};
    __builtin_amdgcn_tensor_load_to_lds(g0, g1, g2, g3, g4, 0);
}

// ---------------------------------------------------------------------------
// conversion / transpose kernels
// ---------------------------------------------------------------------------

__global__ void k_f32_to_bf16(const float* __restrict__ src, __bf16* __restrict__ dst, int n) {
    int i = blockIdx.x * 256 + threadIdx.x;
    if (i < n) dst[i] = f2bf(src[i]);
}

// x: [B][C][N] f32  ->  xT: [B][N][C] bf16
__global__ void k_transpose_x(const float* __restrict__ x, __bf16* __restrict__ xT) {
    int i = blockIdx.x * 256 + threadIdx.x;           // B*C*N threads
    int n = i % NN;
    int c = (i / NN) % CC;
    int b = i / (NN * CC);
    xT[(size_t)b * NN * CC + (size_t)n * CC + c] = f2bf(x[i]);
}

__global__ void k_zero(float* __restrict__ p, int n) {
    int i = blockIdx.x * 256 + threadIdx.x;
    if (i < n) p[i] = 0.0f;
}

// ---------------------------------------------------------------------------
// projection GEMM:  Y = W(MxK=256) @ X(K=256 x N) + bias,  per batch
// MODE 0: store Y transposed [B][N][M]  (q,k)   MODE 1: natural [B][M][N] (v)
// grid: (B * N/64, M/32), block 256
// ---------------------------------------------------------------------------
template <int MODE>
__global__ void k_proj(const __bf16* __restrict__ Wb, const __bf16* __restrict__ xT,
                       const float* __restrict__ bias, __bf16* __restrict__ out, int M) {
    int tid = threadIdx.x, lane = tid & 31, w = tid >> 5;
    int h = (lane >> 4) & 1;
    int nb = blockIdx.x % (NN / 64);
    int b  = blockIdx.x / (NN / 64);
    int o0 = blockIdx.y * 32 + (w & 1) * 16;
    int n0 = nb * 64 + (w >> 1) * 16;
    const __bf16* xb = xT + (size_t)b * NN * CC;

    int col = n0 + (lane & 15);
    int row = o0 + (lane & 15);

    v8f acc;
#pragma unroll
    for (int r = 0; r < 8; ++r) acc[r] = bias[o0 + r + 8 * h];

    for (int k0 = 0; k0 < CC; k0 += 32) {
        v16bf a  = load_fragA(Wb, CC, row, k0, lane);
        v16bf bf = load_fragB(xb, CC, col, k0, lane);
        acc = wmma_bf16(a, bf, acc);
    }

#pragma unroll
    for (int r = 0; r < 8; ++r) {
        int o = o0 + r + 8 * h;
        if (MODE == 0)
            out[(size_t)b * NN * M + (size_t)col * M + o] = f2bf(acc[r]);
        else
            out[(size_t)b * M * NN + (size_t)o * NN + col] = f2bf(acc[r]);
    }
}

// ---------------------------------------------------------------------------
// flash attention:  attT[b][i][c] = sum_j softmax_j(q.k) * v[c][j]
// qT,kT : [B][N][32] bf16 ; v : [B][256][N] bf16 ; attT : [B][N][256] bf16
//
// Transposed score trick: St = K_tile x Q_tile so each lane owns a whole
// softmax row, and P stays lane-resident as the next A fragment.  V tiles
// (256x32 bf16 = 16KB) are double-buffered in LDS via the Tensor Data Mover
// (one descriptor per tile, wave 0 issues, TENSORcnt-synchronized), with the
// next tile's DMA overlapping the current QK^T + softmax.  K fragments are
// register double-buffered.  grid (N/16, B), block 256.
// ---------------------------------------------------------------------------
__global__ void k_attn(const __bf16* __restrict__ qT, const __bf16* __restrict__ kT,
                       const __bf16* __restrict__ v, __bf16* __restrict__ attT) {
    __shared__ __attribute__((aligned(64))) __bf16 sV[2][CC][32];   // 2 x 16 KB
    __shared__ __attribute__((aligned(64))) float  sAl[8][16];
    __shared__ __attribute__((aligned(64))) float  sLl[8][16];

    int tid = threadIdx.x, lane = tid & 31, w = tid >> 5;
    int h  = (lane >> 4) & 1;
    int jl = lane & 15;
    int i0 = blockIdx.x * 16;
    int b  = blockIdx.y;
    int c0 = w * 32;

    const __bf16* qb = qT + (size_t)b * NN * CQKD;
    const __bf16* kb = kT + (size_t)b * NN * CQKD;
    const __bf16* vb = v  + (size_t)b * CC * NN;

    // Q as B-operand (cols i = i0..i0+15), constant over the j loop
    v16bf bQ = load_fragB(qb, CQKD, i0 + jl, 0, lane);

    float m = -3.0e38f, l = 0.0f;   // per-lane softmax state for column i0+jl
    v8f O0 = {}, O1 = {};

    unsigned svbase = (unsigned)(size_t)(&sV[0][0][0]);

    // TDM: kick off V tile 0 into buffer 0 (single DMA descriptor, wave 0)
    if (w == 0) tdm_load_v_tile(svbase, vb);

    // K fragments for tile 0 (register pipeline)
    v16bf aK0 = load_fragA(kb, CQKD, jl,      0, lane);
    v16bf aK1 = load_fragA(kb, CQKD, 16 + jl, 0, lane);

    int buf = 0;
    for (int j0 = 0; j0 < NN; j0 += 32) {
        // pipeline: issue next K-tile fragment loads before consuming current
        int jn = (j0 + 32 < NN) ? (j0 + 32) : j0;
        v16bf nK0 = load_fragA(kb, CQKD, jn + jl,      0, lane);
        v16bf nK1 = load_fragA(kb, CQKD, jn + 16 + jl, 0, lane);
        int jp = (j0 + 96 < NN) ? (j0 + 96) : 0;
        __builtin_prefetch(kb + (size_t)jp * CQKD, 0, 1);

        v8f z = {};
        v8f St0 = wmma_bf16(aK0, bQ, z);   // rows j0 + r + 8h, col i0+jl
        v8f St1 = wmma_bf16(aK1, bQ, z);   // rows j0 + 16 + r + 8h

        // softmax row stats: 16 values in-lane + one xor-16 fold
        float vmax = fmaxf(St0[0], St1[0]);
#pragma unroll
        for (int r = 1; r < 8; ++r) vmax = fmaxf(vmax, fmaxf(St0[r], St1[r]));
        vmax = fmaxf(vmax, __shfl_xor(vmax, 16, 32));
        float nm    = fmaxf(m, vmax);
        float alpha = __expf(m - nm);
        m = nm;

        float p0[8], p1[8], ps = 0.0f;
#pragma unroll
        for (int r = 0; r < 8; ++r) {
            p0[r] = __expf(St0[r] - nm);
            p1[r] = __expf(St1[r] - nm);
            ps += p0[r] + p1[r];
        }
        ps += __shfl_xor(ps, 16, 32);
        l = l * alpha + ps;

        // broadcast per-row alpha through per-wave LDS (rows of O live across lanes)
        if (h == 0) sAl[w][jl] = alpha;
        v8f al = *(const v8f*)(&sAl[w][8 * h]);
#pragma unroll
        for (int r = 0; r < 8; ++r) { O0[r] *= al[r]; O1[r] *= al[r]; }

        // P already sits in the A-fragment layout for this lane: just pack bf16
        v16bf aP;
#pragma unroll
        for (int r = 0; r < 8; ++r) { aP[r] = f2bf(p0[r]); aP[8 + r] = f2bf(p1[r]); }

        // all waves finished reading sV[buf^1] (previous tile) before overwrite
        __syncthreads();
        if (w == 0) {
            if (j0 + 32 < NN) {
                tdm_load_v_tile(svbase + (unsigned)(buf ^ 1) * (unsigned)(CC * 32 * 2),
                                vb + (j0 + 32));
                __builtin_amdgcn_s_wait_tensorcnt(1);  // tile j0 done, next in flight
            } else {
                __builtin_amdgcn_s_wait_tensorcnt(0);
            }
        }
        __syncthreads();   // V tile for this step visible to all waves

        // B-operand fragments for V from LDS (K = j contiguous per channel row)
        v16bf bV0 = frag_from_contig(&sV[buf][c0 + jl][8 * h],
                                     &sV[buf][c0 + jl][16 + 8 * h]);
        v16bf bV1 = frag_from_contig(&sV[buf][c0 + 16 + jl][8 * h],
                                     &sV[buf][c0 + 16 + jl][16 + 8 * h]);
        O0 = wmma_bf16(aP, bV0, O0);
        O1 = wmma_bf16(aP, bV1, O1);

        aK0 = nK0; aK1 = nK1;
        buf ^= 1;
    }

    // final 1/l per output row, broadcast through per-wave LDS
    if (h == 0) sLl[w][jl] = l;
    v8f lr = *(const v8f*)(&sLl[w][8 * h]);

#pragma unroll
    for (int r = 0; r < 8; ++r) {
        float inv = 1.0f / lr[r];
        int n = i0 + r + 8 * h;
        attT[((size_t)b * NN + n) * CC + c0 + jl]      = f2bf(O0[r] * inv);
        attT[((size_t)b * NN + n) * CC + c0 + 16 + jl] = f2bf(O1[r] * inv);
    }
}

// ---------------------------------------------------------------------------
// combine: comb[o][n] = Wc[o,0:256]@x1 + Wc[o,256:512]@att + bc[o]
//          out[b][n] += sum_o |comb[o][n]|   (f32 atomics, out pre-zeroed)
// attT operand tile (16 x 256 bf16 = 8KB) is staged in LDS via
// global_load_async_to_lds_b128 (ASYNCcnt) and shared by all 8 waves.
// grid (N/16, B), block 256: wave w owns output rows [32w, 32w+32)
// ---------------------------------------------------------------------------
__global__ void k_combine(const __bf16* __restrict__ Wcb, const __bf16* __restrict__ xT1,
                          const __bf16* __restrict__ attT, const float* __restrict__ bc,
                          float* __restrict__ out) {
    __shared__ __attribute__((aligned(64))) __bf16 sAtt[16][CC];    // 8 KB

    int tid = threadIdx.x, lane = tid & 31, w = tid >> 5;
    int h  = (lane >> 4) & 1;
    int jl = lane & 15;
    int n0 = blockIdx.x * 16;
    int b  = blockIdx.y;
    int col = n0 + jl;
    int o0  = w * 32;

    const __bf16* xb = xT1  + (size_t)b * NN * CC;
    const __bf16* ab = attT + (size_t)b * NN * CC;

    // async-stage the att tile: thread t owns 32B of row (t>>4)
    {
        int rn = tid >> 4, cseg = tid & 15;
        const __bf16* g = ab + (size_t)(n0 + rn) * CC + cseg * 16;
        unsigned d = (unsigned)(size_t)(&sAtt[0][0])
                     + (unsigned)rn * (unsigned)(CC * 2) + (unsigned)cseg * 32u;
        asm volatile("global_load_async_to_lds_b128 %0, %1, off"
                     :: "v"(d), "v"(g) : "memory");
        asm volatile("global_load_async_to_lds_b128 %0, %1, off offset:16"
                     :: "v"(d), "v"(g) : "memory");
    }

    v8f acc0, acc1;
#pragma unroll
    for (int r = 0; r < 8; ++r) {
        acc0[r] = bc[o0 + r + 8 * h];
        acc1[r] = bc[o0 + 16 + r + 8 * h];
    }

    asm volatile("s_wait_asynccnt 0" ::: "memory");
    __syncthreads();

    for (int k0 = 0; k0 < CC; k0 += 32) {
        v16bf bx = load_fragB(xb, CC, col, k0, lane);
        v16bf a0 = load_fragA(Wcb, 2 * CC, o0 + jl,      k0, lane);
        v16bf a1 = load_fragA(Wcb, 2 * CC, o0 + 16 + jl, k0, lane);
        acc0 = wmma_bf16(a0, bx, acc0);
        acc1 = wmma_bf16(a1, bx, acc1);

        v16bf ba = frag_from_contig(&sAtt[jl][k0 + 16 * h],
                                    &sAtt[jl][k0 + 16 * h] + 8);
        v16bf a2 = load_fragA(Wcb, 2 * CC, o0 + jl,      CC + k0, lane);
        v16bf a3 = load_fragA(Wcb, 2 * CC, o0 + 16 + jl, CC + k0, lane);
        acc0 = wmma_bf16(a2, ba, acc0);
        acc1 = wmma_bf16(a3, ba, acc1);
    }

    float s = 0.0f;
#pragma unroll
    for (int r = 0; r < 8; ++r) s += fabsf(acc0[r]) + fabsf(acc1[r]);
    s += __shfl_xor(s, 16, 32);               // fold the two row-halves
    if (lane < 16) atomicAdd(&out[(size_t)b * NN + col], s);
}

// ---------------------------------------------------------------------------
// host launcher
// ---------------------------------------------------------------------------
extern "C" void kernel_launch(void* const* d_in, const int* in_sizes, int n_in,
                              void* d_out, int out_size, void* d_ws, size_t ws_size,
                              hipStream_t stream) {
    (void)in_sizes; (void)n_in; (void)out_size; (void)ws_size;

    const float* x1 = (const float*)d_in[0];
    const float* x2 = (const float*)d_in[1];
    const float* Wq = (const float*)d_in[2];
    const float* bq = (const float*)d_in[3];
    const float* Wk = (const float*)d_in[4];
    const float* bk = (const float*)d_in[5];
    const float* Wv = (const float*)d_in[6];
    const float* bv = (const float*)d_in[7];
    const float* Wc = (const float*)d_in[8];
    const float* bc = (const float*)d_in[9];
    float* out = (float*)d_out;

    char* ws = (char*)d_ws;
    size_t off = 0;
    auto alloc = [&](size_t bytes) -> void* {
        void* p = ws + off;
        off += (bytes + 255) & ~(size_t)255;
        return p;
    };

    __bf16* Wqb  = (__bf16*)alloc((size_t)CQKD * CC * 2);
    __bf16* Wkb  = (__bf16*)alloc((size_t)CQKD * CC * 2);
    __bf16* Wvb  = (__bf16*)alloc((size_t)CC * CC * 2);
    __bf16* Wcb  = (__bf16*)alloc((size_t)CC * 2 * CC * 2);
    __bf16* xT1  = (__bf16*)alloc((size_t)BB * NN * CC * 2);
    __bf16* xT2  = (__bf16*)alloc((size_t)BB * NN * CC * 2);
    __bf16* qT1  = (__bf16*)alloc((size_t)BB * NN * CQKD * 2);
    __bf16* kT1  = (__bf16*)alloc((size_t)BB * NN * CQKD * 2);
    __bf16* qT2  = (__bf16*)alloc((size_t)BB * NN * CQKD * 2);
    __bf16* kT2  = (__bf16*)alloc((size_t)BB * NN * CQKD * 2);
    __bf16* v1   = (__bf16*)alloc((size_t)BB * CC * NN * 2);
    __bf16* v2   = (__bf16*)alloc((size_t)BB * CC * NN * 2);
    __bf16* at1  = (__bf16*)alloc((size_t)BB * NN * CC * 2);
    __bf16* at2  = (__bf16*)alloc((size_t)BB * NN * CC * 2);

    // --- weight conversion -------------------------------------------------
    k_f32_to_bf16<<<(CQKD * CC + 255) / 256, 256, 0, stream>>>(Wq, Wqb, CQKD * CC);
    k_f32_to_bf16<<<(CQKD * CC + 255) / 256, 256, 0, stream>>>(Wk, Wkb, CQKD * CC);
    k_f32_to_bf16<<<(CC * CC + 255) / 256, 256, 0, stream>>>(Wv, Wvb, CC * CC);
    k_f32_to_bf16<<<(CC * 2 * CC + 255) / 256, 256, 0, stream>>>(Wc, Wcb, CC * 2 * CC);

    // --- activations -> bf16 transposed ------------------------------------
    int xtot = BB * CC * NN;
    k_transpose_x<<<xtot / 256, 256, 0, stream>>>(x1, xT1);
    k_transpose_x<<<xtot / 256, 256, 0, stream>>>(x2, xT2);

    // --- projections --------------------------------------------------------
    dim3 gqk(BB * (NN / 64), CQKD / 32);
    dim3 gv (BB * (NN / 64), CC / 32);
    k_proj<0><<<gqk, 256, 0, stream>>>(Wqb, xT1, bq, qT1, CQKD);
    k_proj<0><<<gqk, 256, 0, stream>>>(Wkb, xT1, bk, kT1, CQKD);
    k_proj<1><<<gv,  256, 0, stream>>>(Wvb, xT1, bv, v1,  CC);
    k_proj<0><<<gqk, 256, 0, stream>>>(Wqb, xT2, bq, qT2, CQKD);
    k_proj<0><<<gqk, 256, 0, stream>>>(Wkb, xT2, bk, kT2, CQKD);
    k_proj<1><<<gv,  256, 0, stream>>>(Wvb, xT2, bv, v2,  CC);

    // --- flash attention (branch1 uses v2, branch2 uses v1) -----------------
    dim3 ga(NN / 16, BB);
    k_attn<<<ga, 256, 0, stream>>>(qT1, kT1, v2, at1);
    k_attn<<<ga, 256, 0, stream>>>(qT2, kT2, v1, at2);

    // --- combine (both use x1, per reference) -------------------------------
    int osz = 2 * BB * NN;
    k_zero<<<(osz + 255) / 256, 256, 0, stream>>>(out, osz);
    dim3 gc(NN / 16, BB);
    k_combine<<<gc, 256, 0, stream>>>(Wcb, xT1, at1, bc, out);
    k_combine<<<gc, 256, 0, stream>>>(Wcb, xT1, at2, bc, out + (size_t)BB * NN);
}
